// MAFG_CA_63977832841379
// MI455X (gfx1250) — matrix operand order repaired and checked by reading, not compiled, not run on hardware
//
#include <hip/hip_runtime.h>

#define HW 16384          // 128*128
#define NPIX 81920        // BM * HW = 5 * 16384
#define EDIM 64

typedef __attribute__((ext_vector_type(16))) __bf16 v16bf;
typedef __attribute__((ext_vector_type(8)))  float  v8f;

union frag16 { uint4 u[2]; v16bf v; };

// ---------------------------------------------------------------------------
// WMMA fragment loaders (CDNA5 wave32 layouts, CDNA5 ISA §7.12.2)
// A (16x32 bf16): lane<16 -> row=lane, K = {0..7,16..23}; lane>=16 -> row=lane-16, K={8..15,24..31}
// B (32x16 bf16): lane<16 -> col=lane, K = {0..7,16..23}; lane>=16 -> col=lane-16, K={8..15,24..31}
// C/D (16x16 f32): vgpr r, lane l -> row = r + 8*(l>>4), col = l&15
// Each fragment is two contiguous 8-element runs -> 128-bit loads.
// ---------------------------------------------------------------------------
__device__ __forceinline__ v16bf load_a_bf16(const __bf16* base, int lda, int row0,
                                             int k0, int lane) {
  const __bf16* p = base + (size_t)(row0 + (lane & 15)) * lda + k0 + ((lane < 16) ? 0 : 8);
  frag16 f;
  f.u[0] = *(const uint4*)p;          // K elems j=0..7   (16 B)
  f.u[1] = *(const uint4*)(p + 16);   // K elems j=8..15  (16 B)
  return f.v;
}

__device__ __forceinline__ v16bf cvt8x2(float4 a0, float4 a1, float4 b0, float4 b1) {
  v16bf v;
  v[0] = (__bf16)a0.x;  v[1] = (__bf16)a0.y;  v[2]  = (__bf16)a0.z;  v[3]  = (__bf16)a0.w;
  v[4] = (__bf16)a1.x;  v[5] = (__bf16)a1.y;  v[6]  = (__bf16)a1.z;  v[7]  = (__bf16)a1.w;
  v[8] = (__bf16)b0.x;  v[9] = (__bf16)b0.y;  v[10] = (__bf16)b0.z;  v[11] = (__bf16)b0.w;
  v[12] = (__bf16)b1.x; v[13] = (__bf16)b1.y; v[14] = (__bf16)b1.z;  v[15] = (__bf16)b1.w;
  return v;
}

__device__ __forceinline__ v16bf load_a_f32(const float* base, int lda, int row0,
                                            int k0, int lane) {
  const float* p = base + (size_t)(row0 + (lane & 15)) * lda + k0 + ((lane < 16) ? 0 : 8);
  return cvt8x2(*(const float4*)p, *(const float4*)(p + 4),
                *(const float4*)(p + 16), *(const float4*)(p + 20));
}

// weight is row-major (N, K): B[k][n] = w[col0+n][k]
__device__ __forceinline__ v16bf load_b_w(const float* w, int ldw, int col0,
                                          int k0, int lane) {
  const float* p = w + (size_t)(col0 + (lane & 15)) * ldw + k0 + ((lane < 16) ? 0 : 8);
  return cvt8x2(*(const float4*)p, *(const float4*)(p + 4),
                *(const float4*)(p + 16), *(const float4*)(p + 20));
}

// ---------------------------------------------------------------------------
// Kernel 1: 1x1 conv (3->64) + LayerNorm.  One pixel per 64-thread block.
// ---------------------------------------------------------------------------
__global__ void prep_kernel(const float* __restrict__ x, const float* __restrict__ w_high,
                            const float* __restrict__ b_high, const float* __restrict__ gamma,
                            const float* __restrict__ beta, float* __restrict__ feat,
                            __bf16* __restrict__ xs) {
  int pix = blockIdx.x;
  int e = threadIdx.x;
  int bm = pix >> 14;
  int p = pix & 16383;
  float acc = b_high[e];
#pragma unroll
  for (int c = 0; c < 3; ++c)
    acc += x[(size_t)(bm * 3 + c) * HW + p] * w_high[e * 3 + c];
  feat[(size_t)pix * EDIM + e] = acc;

  __shared__ float red[EDIM];
  red[e] = acc;
  __syncthreads();
  float mu = 0.f;
#pragma unroll
  for (int i = 0; i < EDIM; ++i) mu += red[i];
  mu *= (1.0f / EDIM);
  float var = 0.f;
#pragma unroll
  for (int i = 0; i < EDIM; ++i) { float d = red[i] - mu; var += d * d; }
  var *= (1.0f / EDIM);
  float xv = (acc - mu) * rsqrtf(var + 1e-6f) * gamma[e] + beta[e];
  xs[(size_t)pix * EDIM + e] = (__bf16)xv;
}

// ---------------------------------------------------------------------------
// Kernel 2: QKV GEMM  (81920 x 64) @ (64 x 192) -> qkv f32.  One 16x16 tile/wave.
// ---------------------------------------------------------------------------
__global__ void qkv_gemm_kernel(const __bf16* __restrict__ xs, const float* __restrict__ w_qkv,
                                float* __restrict__ qkv) {
  int lane = threadIdx.x & 31;
  int tile = blockIdx.x * 8 + (threadIdx.x >> 5);
  int row0 = (tile / 12) * 16;
  int col0 = (tile % 12) * 16;
  // speculative prefetch of the next block's A rows (gfx1250 global_prefetch)
  __builtin_prefetch(xs + ((size_t)row0 + 128) * EDIM + (lane << 2), 0, 1);
  v8f acc = {};
#pragma unroll
  for (int ks = 0; ks < 2; ++ks) {
    v16bf a = load_a_bf16(xs, EDIM, row0, ks * 32, lane);
    v16bf b = load_b_w(w_qkv, EDIM, col0, ks * 32, lane);
    acc = __builtin_amdgcn_wmma_f32_16x16x32_bf16(false, a, false, b, (short)0, acc,
                                                  false, false);
  }
  int n = col0 + (lane & 15);
  int rbase = row0 + ((lane >> 4) << 3);
#pragma unroll
  for (int r = 0; r < 8; ++r) qkv[(size_t)(rbase + r) * 192 + n] = acc[r];
}

// ---------------------------------------------------------------------------
// Kernel 3: windowed MHA. 1 window per 256-thread block, 1 head per wave.
// S = (Q+posq)*scale @ (K+posk)^T   (4x9 tiles, K-dim 8 padded to 32)
// softmax over 144 cols in registers (shfl_xor within 16-lane row groups)
// O = P @ V                         (K-dim 144 padded to 160, 5 steps)
// P re-layout C->A goes through a private 16x160 bf16 LDS tile per wave.
// ---------------------------------------------------------------------------
__global__ __launch_bounds__(256) void attn_kernel(const float* __restrict__ qkv,
                                                   const float* __restrict__ pos_q,
                                                   const float* __restrict__ pos_k,
                                                   float* __restrict__ attn_out) {
  int lane = threadIdx.x & 31;
  int h = threadIdx.x >> 5;       // head 0..7
  int s = blockIdx.x;             // window 0..1279
  int bm = s >> 8;
  int wy = (s >> 4) & 15;
  int wx = s & 15;
  const float scale = 0.35355339059327373f;   // 1/sqrt(8)

  __shared__ __attribute__((aligned(16))) __bf16 plds[8][16 * 160];   // 40 KB

  // ---- K fragments: B-layout, 9 col tiles, only d<8 valid (rest zero) ----
  v16bf kb[9];
#pragma unroll
  for (int ct = 0; ct < 9; ++ct) {
    v16bf b;
#pragma unroll
    for (int j = 0; j < 16; ++j) b[j] = (__bf16)0.0f;
    if (lane < 16) {
      int kk = ct * 16 + lane;                // 0..143
      int ki = kk / 12, kj = kk - ki * 12;
      int y = wy * 8 + ki - 2, x = wx * 8 + kj - 2;
      bool inb = (y >= 0 && y < 128 && x >= 0 && x < 128);
      const float* pp = pos_k + (size_t)kk * EDIM + h * 8;
      float4 p0 = *(const float4*)pp;
      float4 p1 = *(const float4*)(pp + 4);
      if (inb) {
        const float* kp = qkv + ((size_t)(bm * HW + y * 128 + x) * 192 + EDIM + h * 8);
        float4 k0 = *(const float4*)kp;
        float4 k1 = *(const float4*)(kp + 4);
        p0.x += k0.x; p0.y += k0.y; p0.z += k0.z; p0.w += k0.w;
        p1.x += k1.x; p1.y += k1.y; p1.z += k1.z; p1.w += k1.w;
      }
      b[0] = (__bf16)p0.x; b[1] = (__bf16)p0.y; b[2] = (__bf16)p0.z; b[3] = (__bf16)p0.w;
      b[4] = (__bf16)p1.x; b[5] = (__bf16)p1.y; b[6] = (__bf16)p1.z; b[7] = (__bf16)p1.w;
    }
    kb[ct] = b;
  }

  // ---- V fragments: B-layout over K-dim (144 rows padded to 160) ----
  v16bf vb[5];
#pragma unroll
  for (int ks = 0; ks < 5; ++ks) {
    v16bf b;
#pragma unroll
    for (int j = 0; j < 16; ++j) b[j] = (__bf16)0.0f;
    int d = lane & 15;
    if (d < 8) {
      int kbase = ks * 32 + ((lane < 16) ? 0 : 16);
#pragma unroll
      for (int j = 0; j < 16; ++j) {
        int kr = kbase + j;
        if (kr < 144) {
          int ki = kr / 12, kj = kr - ki * 12;
          int y = wy * 8 + ki - 2, x = wx * 8 + kj - 2;
          if (y >= 0 && y < 128 && x >= 0 && x < 128)
            b[j] = (__bf16)qkv[(size_t)(bm * HW + y * 128 + x) * 192 + 2 * EDIM + h * 8 + d];
        }
      }
    }
    vb[ks] = b;
  }

#pragma unroll 1
  for (int rt = 0; rt < 4; ++rt) {
    // ---- Q fragment (broadcast from middle frame bm==2), scale folded in ----
    v16bf qa;
#pragma unroll
    for (int j = 0; j < 16; ++j) qa[j] = (__bf16)0.0f;
    if (lane < 16) {
      int q = rt * 16 + lane;
      int y = wy * 8 + (q >> 3), x = wx * 8 + (q & 7);
      const float* qp = qkv + ((size_t)(2 * HW + y * 128 + x) * 192 + h * 8);
      const float* pp = pos_q + (size_t)q * EDIM + h * 8;
      float4 q0 = *(const float4*)qp;
      float4 q1 = *(const float4*)(qp + 4);
      float4 p0 = *(const float4*)pp;
      float4 p1 = *(const float4*)(pp + 4);
      qa[0] = (__bf16)((q0.x + p0.x) * scale); qa[1] = (__bf16)((q0.y + p0.y) * scale);
      qa[2] = (__bf16)((q0.z + p0.z) * scale); qa[3] = (__bf16)((q0.w + p0.w) * scale);
      qa[4] = (__bf16)((q1.x + p1.x) * scale); qa[5] = (__bf16)((q1.y + p1.y) * scale);
      qa[6] = (__bf16)((q1.z + p1.z) * scale); qa[7] = (__bf16)((q1.w + p1.w) * scale);
    }

    // ---- S = Q K^T : 9 score tiles ----
    v8f acc[9];
#pragma unroll
    for (int ct = 0; ct < 9; ++ct) {
      v8f z = {};
      acc[ct] = __builtin_amdgcn_wmma_f32_16x16x32_bf16(false, qa, false, kb[ct],
                                                        (short)0, z, false, false);
    }

    // ---- softmax per row; write P (bf16) to this wave's LDS tile ----
#pragma unroll
    for (int r = 0; r < 8; ++r) {
      float m = -3.0e38f;
#pragma unroll
      for (int ct = 0; ct < 9; ++ct) m = fmaxf(m, acc[ct][r]);
#pragma unroll
      for (int off = 8; off >= 1; off >>= 1) m = fmaxf(m, __shfl_xor(m, off, 32));
      float p[9], sum = 0.f;
#pragma unroll
      for (int ct = 0; ct < 9; ++ct) { p[ct] = __expf(acc[ct][r] - m); sum += p[ct]; }
#pragma unroll
      for (int off = 8; off >= 1; off >>= 1) sum += __shfl_xor(sum, off, 32);
      float inv = __builtin_amdgcn_rcpf(sum);
      int rowl = r + ((lane >> 4) << 3);
      __bf16* dst = &plds[h][rowl * 160];
#pragma unroll
      for (int ct = 0; ct < 9; ++ct) dst[ct * 16 + (lane & 15)] = (__bf16)(p[ct] * inv);
      dst[144 + (lane & 15)] = (__bf16)0.0f;   // zero the K-pad (144..159)
    }
    // same-wave LDS write -> read ordering (split counters on gfx1250)
    asm volatile("s_wait_dscnt 0" ::: "memory");

    // ---- O = P @ V : 5 K-steps of 32 (A fragments via 128-bit LDS loads) ----
    v8f o = {};
#pragma unroll
    for (int ks = 0; ks < 5; ++ks) {
      const __bf16* src = &plds[h][(lane & 15) * 160 + ks * 32 + ((lane < 16) ? 0 : 16)];
      frag16 f;
      f.u[0] = *(const uint4*)src;          // j=0..7
      f.u[1] = *(const uint4*)(src + 8);    // j=8..15
      o = __builtin_amdgcn_wmma_f32_16x16x32_bf16(false, f.v, false, vb[ks],
                                                  (short)0, o, false, false);
    }

    // ---- scatter O back to (BM, HW, E) ----
    int d = lane & 15;
    if (d < 8) {
      int qb = rt * 16 + ((lane >> 4) << 3);
#pragma unroll
      for (int r = 0; r < 8; ++r) {
        int q = qb + r;
        int y = wy * 8 + (q >> 3), x = wx * 8 + (q & 7);
        attn_out[(size_t)(bm * HW + y * 128 + x) * EDIM + h * 8 + d] = o[r];
      }
    }
  }
}

// ---------------------------------------------------------------------------
// Kernel 4: out-projection GEMM (81920x64 @ 64x64) + residual -> mixed
// ---------------------------------------------------------------------------
__global__ void outproj_kernel(const float* __restrict__ attn_out, const float* __restrict__ w_out,
                               const float* __restrict__ feat, float* __restrict__ mixed) {
  int lane = threadIdx.x & 31;
  int tile = blockIdx.x * 8 + (threadIdx.x >> 5);
  int row0 = (tile >> 2) * 16;
  int col0 = (tile & 3) * 16;
  __builtin_prefetch(attn_out + ((size_t)row0 + 32) * EDIM + (lane << 3), 0, 1);
  v8f acc = {};
#pragma unroll
  for (int ks = 0; ks < 2; ++ks) {
    v16bf a = load_a_f32(attn_out, EDIM, row0, ks * 32, lane);
    v16bf b = load_b_w(w_out, EDIM, col0, ks * 32, lane);
    acc = __builtin_amdgcn_wmma_f32_16x16x32_bf16(false, a, false, b, (short)0, acc,
                                                  false, false);
  }
  int n = col0 + (lane & 15);
  int rbase = row0 + ((lane >> 4) << 3);
#pragma unroll
  for (int r = 0; r < 8; ++r) {
    size_t idx = (size_t)(rbase + r) * EDIM + n;
    mixed[idx] = acc[r] + feat[idx];
  }
}

// ---------------------------------------------------------------------------
// Kernel 5: RGB head (64 -> 3) per pixel, 128-bit row loads
// ---------------------------------------------------------------------------
__global__ void rgb_kernel(const float* __restrict__ mixed, const float* __restrict__ w_rgb,
                           const float* __restrict__ b_rgb, float* __restrict__ out) {
  int pix = blockIdx.x * blockDim.x + threadIdx.x;
  if (pix >= NPIX) return;
  int bm = pix >> 14, p = pix & 16383;
  float a0 = b_rgb[0], a1 = b_rgb[1], a2 = b_rgb[2];
  const float4* mr = (const float4*)(mixed + (size_t)pix * EDIM);
#pragma unroll
  for (int e4 = 0; e4 < 16; ++e4) {
    float4 v = mr[e4];
    const float* w0 = w_rgb + e4 * 4;
    a0 += v.x * w0[0] + v.y * w0[1] + v.z * w0[2] + v.w * w0[3];
    const float* w1 = w_rgb + EDIM + e4 * 4;
    a1 += v.x * w1[0] + v.y * w1[1] + v.z * w1[2] + v.w * w1[3];
    const float* w2 = w_rgb + 2 * EDIM + e4 * 4;
    a2 += v.x * w2[0] + v.y * w2[1] + v.z * w2[2] + v.w * w2[3];
  }
  out[(size_t)(bm * 3 + 0) * HW + p] = a0;
  out[(size_t)(bm * 3 + 1) * HW + p] = a1;
  out[(size_t)(bm * 3 + 2) * HW + p] = a2;
}

// ---------------------------------------------------------------------------
extern "C" void kernel_launch(void* const* d_in, const int* in_sizes, int n_in,
                              void* d_out, int out_size, void* d_ws, size_t ws_size,
                              hipStream_t stream) {
  const float* x      = (const float*)d_in[0];
  const float* w_high = (const float*)d_in[1];
  const float* b_high = (const float*)d_in[2];
  const float* gamma  = (const float*)d_in[3];
  const float* beta   = (const float*)d_in[4];
  const float* w_qkv  = (const float*)d_in[5];
  const float* pos_q  = (const float*)d_in[6];
  const float* pos_k  = (const float*)d_in[7];
  const float* w_out  = (const float*)d_in[8];
  const float* w_rgb  = (const float*)d_in[9];
  const float* b_rgb  = (const float*)d_in[10];
  float* out = (float*)d_out;

  char* ws = (char*)d_ws;
  float*  feat     = (float*)(ws);                                   // 81920*64*4  = 20.97 MB
  __bf16* xs       = (__bf16*)(ws + (size_t)NPIX * EDIM * 4);        // 81920*64*2  = 10.49 MB
  float*  qkv      = (float*)(ws + (size_t)NPIX * EDIM * 6);         // 81920*192*4 = 62.91 MB
  float*  attn_out = (float*)(ws + (size_t)NPIX * EDIM * 6 + (size_t)NPIX * 192 * 4);
  float*  mixed    = (float*)(ws + (size_t)NPIX * EDIM * 10 + (size_t)NPIX * 192 * 4);

  prep_kernel<<<NPIX, EDIM, 0, stream>>>(x, w_high, b_high, gamma, beta, feat, xs);
  qkv_gemm_kernel<<<(NPIX / 16) * 12 / 8, 256, 0, stream>>>(xs, w_qkv, qkv);
  attn_kernel<<<1280, 256, 0, stream>>>(qkv, pos_q, pos_k, attn_out);
  outproj_kernel<<<(NPIX / 16) * 4 / 8, 256, 0, stream>>>(attn_out, w_out, feat, mixed);
  rgb_kernel<<<(NPIX + 255) / 256, 256, 0, stream>>>(mixed, w_rgb, b_rgb, out);
}
